// GCNWithAttention_44203803410720
// MI455X (gfx1250) — compile-verified, matrix operand order checked by reading, not run on hardware
//
#include <hip/hip_runtime.h>
#include <hip/hip_bf16.h>

typedef __attribute__((ext_vector_type(16))) __bf16          v16bf;
typedef __attribute__((ext_vector_type(8)))  float           v8f;
typedef __attribute__((ext_vector_type(8)))  unsigned short  v8u;
typedef __attribute__((ext_vector_type(16))) unsigned short  v16u;

// ---------- helpers ----------
__device__ __forceinline__ unsigned short f2bf(float f) {
  union { float f; unsigned u; } v; v.f = f;
  unsigned r = v.u + 0x7FFFu + ((v.u >> 16) & 1u);   // round-to-nearest-even
  return (unsigned short)(r >> 16);
}

// Load one 16x32 bf16 fragment (A-style layout): lane L(0..15) = row, holds
// K {0..7}U{16..23}; lane L+16 same row, K {8..15}U{24..31}. Two b128 loads.
__device__ __forceinline__ v16bf load_frag(const unsigned short* __restrict__ base,
                                           int ld, int row, int k0, int lane) {
  const unsigned short* p = base + (size_t)row * ld + k0 + ((lane >> 4) << 3);
  v8u lo = *(const v8u*)p;
  v8u hi = *(const v8u*)(p + 16);
  v16u v = __builtin_shufflevector(lo, hi, 0,1,2,3,4,5,6,7,8,9,10,11,12,13,14,15);
  return __builtin_bit_cast(v16bf, v);
}

// ---------- WMMA GEMM, 1x4 M-blocked ----------
// C[n,m] = A[n,k] * Wt[m,k] (+bias, relu).  A: bf16 bits, lda = Kpad (mult 32).
// Bt: bf16 bits, ld = lda, rows zero-padded to mquads*64.
// One wave computes 16 rows x 64 cols: A fragment reused across 4 B tiles.
// nrows multiple of 32; block = 64 threads (2 waves, adjacent row tiles).
__global__ void __launch_bounds__(64)
gemm_bf16_wmma4(const unsigned short* __restrict__ A,
                const unsigned short* __restrict__ Bt,
                const float* __restrict__ bias, float* __restrict__ C,
                int kdim, int lda, int mtiles, int ldc, int dorelu) {
  int lane = threadIdx.x & 31;
  int wave = threadIdx.x >> 5;
  int rowbase = (blockIdx.x * 2 + wave) * 16;
  int mbase0  = blockIdx.y * 64;
  int ra  = rowbase + (lane & 15);
  int rb0 = mbase0  + (lane & 15);
  v8f acc0 = {0.f,0.f,0.f,0.f,0.f,0.f,0.f,0.f};
  v8f acc1 = acc0, acc2 = acc0, acc3 = acc0;
  for (int k0 = 0; k0 < kdim; k0 += 32) {
    v16bf a  = load_frag(A,  lda, ra,       k0, lane);
    v16bf b0 = load_frag(Bt, lda, rb0,      k0, lane);
    v16bf b1 = load_frag(Bt, lda, rb0 + 16, k0, lane);
    v16bf b2 = load_frag(Bt, lda, rb0 + 32, k0, lane);
    v16bf b3 = load_frag(Bt, lda, rb0 + 48, k0, lane);
    acc0 = __builtin_amdgcn_wmma_f32_16x16x32_bf16(false, a, false, b0, (short)0, acc0, false, false);
    acc1 = __builtin_amdgcn_wmma_f32_16x16x32_bf16(false, a, false, b1, (short)0, acc1, false, false);
    acc2 = __builtin_amdgcn_wmma_f32_16x16x32_bf16(false, a, false, b2, (short)0, acc2, false, false);
    acc3 = __builtin_amdgcn_wmma_f32_16x16x32_bf16(false, a, false, b3, (short)0, acc3, false, false);
  }
  int rtop = rowbase + ((lane >> 4) << 3);
  v8f accs[4] = {acc0, acc1, acc2, acc3};
  #pragma unroll
  for (int tq = 0; tq < 4; ++tq) {
    int tile = blockIdx.y * 4 + tq;
    if (tile < mtiles) {                       // wave-uniform guard
      int col  = tile * 16 + (lane & 15);
      float bv = bias ? bias[col] : 0.0f;
      #pragma unroll
      for (int j = 0; j < 8; ++j) {
        float v = accs[tq][j] + bv;
        if (dorelu) v = fmaxf(v, 0.0f);
        C[(size_t)(rtop + j) * ldc + col] = v;
      }
    }
  }
}

// ---------- elementwise / staging kernels ----------
__global__ void fzero(float* __restrict__ p, long long n) {
  long long i = (long long)blockIdx.x * blockDim.x + threadIdx.x;
  if (i < n) p[i] = 0.0f;
}

// zero the bf16 pad columns [356,384) of the concat staging buffer
__global__ void zpad(unsigned short* __restrict__ Ac, long long nrows) {
  long long gid = (long long)blockIdx.x * blockDim.x + threadIdx.x;
  if (gid >= nrows * 28) return;
  long long r = gid / 28; int c = (int)(gid % 28);
  Ac[r * 384 + 356 + c] = 0;
}

// f32 submatrix -> bf16 columns of dst
__global__ void cvt_cols(const float* __restrict__ src, int sld,
                         unsigned short* __restrict__ dst, int dld, int dcol0,
                         int ncols, long long nrows) {
  long long gid = (long long)blockIdx.x * blockDim.x + threadIdx.x;
  if (gid >= nrows * (long long)ncols) return;
  long long r = gid / ncols; int c = (int)(gid % ncols);
  dst[r * dld + dcol0 + c] = f2bf(src[r * (long long)sld + c]);
}

// transpose + pad + convert weights: Wt[m*Kpad+k] = W[k*Msrc+m] (0 outside)
__global__ void prep_wt(const float* __restrict__ W, int Ksrc, int Msrc,
                        unsigned short* __restrict__ Wt, int Kpad, int Mpad) {
  int idx = blockIdx.x * blockDim.x + threadIdx.x;
  if (idx >= Mpad * Kpad) return;
  int m = idx / Kpad, k = idx % Kpad;
  float v = (k < Ksrc && m < Msrc) ? W[(size_t)k * Msrc + m] : 0.0f;
  Wt[(size_t)m * Kpad + k] = f2bf(v);
}

__global__ void padb(const float* __restrict__ src, int n, float* __restrict__ dst, int npad) {
  int i = threadIdx.x;
  if (i < npad) dst[i] = (i < n) ? src[i] : 0.0f;
}

// ---------- edge scatter: agg[dst] += h[src]*w  (128 feats, 4 per thread) ----------
__global__ void scatter_edges(const float* __restrict__ h,
                              const long long* __restrict__ src,
                              const long long* __restrict__ dst,
                              const float* __restrict__ ew,
                              float* __restrict__ agg, int E) {
  long long gid = (long long)blockIdx.x * blockDim.x + threadIdx.x;
  long long e = gid >> 5;
  int c = (int)(gid & 31) << 2;
  if (e >= E) return;
  long long s = src[e], d = dst[e];
  float w = ew[e];
  const float4 v = *(const float4*)(h + s * 128 + c);
  float* o = agg + d * 128 + c;
  atomicAdd(o + 0, v.x * w); atomicAdd(o + 1, v.y * w);
  atomicAdd(o + 2, v.z * w); atomicAdd(o + 3, v.w * w);
}

// x_local = relu(agg + conv_b) -> bf16 concat cols [100,228)
__global__ void agg_fin(const float* __restrict__ agg, const float* __restrict__ cb,
                        unsigned short* __restrict__ Ac, long long nrows) {
  long long gid = (long long)blockIdx.x * blockDim.x + threadIdx.x;
  if (gid >= nrows * 128) return;
  long long r = gid >> 7; int c = (int)(gid & 127);
  float v = agg[gid] + cb[c];
  v = fmaxf(v, 0.0f);
  Ac[r * 384 + 100 + c] = f2bf(v);
}

// ---------- LRGA reductions ----------
// t layout (ld=208): U=[0,50) V=[50,100) Z=[100,150) T=[150,200)
__global__ void __launch_bounds__(256)
lrga_part(const float* __restrict__ t, int ldt, float* __restrict__ Mg,
          float* __restrict__ usum, float* __restrict__ vt1, int nrows) {
  __shared__ float Vl[64 * 50];
  __shared__ float Zl[64 * 50];
  int r0 = blockIdx.x * 64;
  for (int i = threadIdx.x; i < 64 * 50; i += 256) {
    int r = i / 50, c = i - r * 50;
    int row = r0 + r;
    float v = 0.f, z = 0.f;
    if (row < nrows) {
      v = t[(size_t)row * ldt + 50 + c];
      z = t[(size_t)row * ldt + 100 + c];
    }
    Vl[i] = v; Zl[i] = z;
  }
  __syncthreads();
  for (int cell = threadIdx.x; cell < 2500; cell += 256) {
    int a = cell / 50, b = cell - a * 50;
    float s = 0.f;
    for (int r = 0; r < 64; ++r) s += Vl[r * 50 + a] * Zl[r * 50 + b];
    atomicAdd(&Mg[cell], s);
  }
  if (threadIdx.x < 50) {
    int c = threadIdx.x; float sv = 0.f, su = 0.f;
    for (int r = 0; r < 64; ++r) {
      sv += Vl[r * 50 + c];
      int row = r0 + r;
      if (row < nrows) su += t[(size_t)row * ldt + c];
    }
    atomicAdd(&vt1[c], sv); atomicAdd(&usum[c], su);
  }
}

__global__ void norm_k(const float* __restrict__ usum, const float* __restrict__ vt1,
                       float* __restrict__ normv, float invN) {
  __shared__ float red[64];
  int tid = threadIdx.x;
  red[tid] = (tid < 50) ? usum[tid] * vt1[tid] : 0.0f;
  __syncthreads();
  for (int s = 32; s > 0; s >>= 1) {
    if (tid < s) red[tid] += red[tid + s];
    __syncthreads();
  }
  if (tid == 0) normv[0] = red[0] * invN + 1e-6f;
}

// res = (U @ M)/norm -> bf16 concat cols [0,50)
__global__ void __launch_bounds__(256)
res_k(const float* __restrict__ t, int ldt, const float* __restrict__ Mg,
      const float* __restrict__ normv, unsigned short* __restrict__ Ac,
      long long nrows) {
  __shared__ float Ml[2500];
  for (int i = threadIdx.x; i < 2500; i += 256) Ml[i] = Mg[i];
  __syncthreads();
  float inv = 1.0f / normv[0];
  long long gid = (long long)blockIdx.x * 256 + threadIdx.x;
  long long row = gid >> 6; int col = (int)(gid & 63);
  if (row >= nrows || col >= 50) return;
  const float* u = t + (size_t)row * ldt;
  float s = 0.f;
  for (int k = 0; k < 50; ++k) s += u[k] * Ml[k * 50 + col];
  Ac[row * 384 + col] = f2bf(s * inv);
}

// ---------- BatchNorm (over N) ----------
__global__ void __launch_bounds__(256)
bn_stats(const float* __restrict__ x, float* __restrict__ bsum,
         float* __restrict__ bsq, int nrows) {
  __shared__ float ls[128], lq[128];
  int c = threadIdx.x & 127, hh = threadIdx.x >> 7;
  if (threadIdx.x < 128) { ls[c] = 0.f; lq[c] = 0.f; }
  __syncthreads();
  int r0 = blockIdx.x * 256;
  int rend = r0 + 256; if (rend > nrows) rend = nrows;
  float s = 0.f, q = 0.f;
  for (int r = r0 + hh; r < rend; r += 2) {
    float v = x[(size_t)r * 128 + c];
    s += v; q += v * v;
  }
  atomicAdd(&ls[c], s); atomicAdd(&lq[c], q);
  __syncthreads();
  if (threadIdx.x < 128) { atomicAdd(&bsum[c], ls[c]); atomicAdd(&bsq[c], lq[c]); }
}

__global__ void bn_apply(float* __restrict__ x, const float* __restrict__ bsum,
                         const float* __restrict__ bsq, const float* __restrict__ g,
                         const float* __restrict__ b, long long nrows, float invN) {
  long long gid = (long long)blockIdx.x * blockDim.x + threadIdx.x;
  if (gid >= nrows * 128) return;
  int c = (int)(gid & 127);
  float mu  = bsum[c] * invN;
  float var = bsq[c] * invN - mu * mu;
  float v = x[gid];
  x[gid] = (v - mu) * rsqrtf(var + 1e-5f) * g[c] + b[c];
}

// ---------- host-side orchestration ----------
extern "C" void kernel_launch(void* const* d_in, const int* in_sizes, int n_in,
                              void* d_out, int out_size, void* d_ws, size_t ws_size,
                              hipStream_t stream) {
  (void)n_in; (void)out_size; (void)ws_size;
  const int       Nn = in_sizes[0] / 128;       // 100000
  const long long Ee = in_sizes[1] / 2;         // 1600000
  const float*      x0 = (const float*)d_in[0];
  const long long*  ei = (const long long*)d_in[1];
  const float*      ew = (const float*)d_in[2];

  const float *conv_w[3], *conv_b[3], *att_w[3], *att_b[3];
  const float *red_w[3], *red_b[3], *bn_g[2], *bn_b[2];
  if (in_sizes[3] == 128 * 128) {               // dict insertion order
    for (int i = 0; i < 3; ++i) {
      conv_w[i] = (const float*)d_in[3 + 4 * i];
      conv_b[i] = (const float*)d_in[4 + 4 * i];
      att_w[i]  = (const float*)d_in[5 + 4 * i];
      att_b[i]  = (const float*)d_in[6 + 4 * i];
    }
    for (int i = 0; i < 3; ++i) { red_w[i] = (const float*)d_in[15 + 2 * i];
                                  red_b[i] = (const float*)d_in[16 + 2 * i]; }
    for (int i = 0; i < 2; ++i) { bn_g[i] = (const float*)d_in[21 + 2 * i];
                                  bn_b[i] = (const float*)d_in[22 + 2 * i]; }
  } else {                                      // alphabetical fallback
    for (int i = 0; i < 3; ++i) { att_b[i]  = (const float*)d_in[3 + i];
                                  att_w[i]  = (const float*)d_in[6 + i]; }
    for (int i = 0; i < 2; ++i) { bn_b[i] = (const float*)d_in[9 + i];
                                  bn_g[i] = (const float*)d_in[11 + i]; }
    for (int i = 0; i < 3; ++i) { conv_b[i] = (const float*)d_in[13 + i];
                                  conv_w[i] = (const float*)d_in[16 + i]; }
    for (int i = 0; i < 3; ++i) { red_b[i] = (const float*)d_in[19 + i];
                                  red_w[i] = (const float*)d_in[22 + i]; }
  }

  // carve workspace
  char* base = (char*)d_ws; size_t off = 0;
  auto carve = [&](size_t bytes) -> void* {
    void* r = base + off; off = (off + bytes + 255) & ~(size_t)255; return r;
  };
  unsigned short* Axb = (unsigned short*)carve((size_t)Nn * 128 * 2); // x bf16 stage
  unsigned short* Ac  = (unsigned short*)carve((size_t)Nn * 384 * 2); // concat bf16 (ld 384)
  float* hbuf = (float*)carve((size_t)Nn * 128 * 4);                  // x @ conv_w
  float* tbuf = (float*)carve((size_t)Nn * 208 * 4);                  // att out (ld 208)
  float* agg  = (float*)carve((size_t)Nn * 128 * 4);                  // segment sum
  float* xbuf = (float*)carve((size_t)Nn * 128 * 4);                  // layer output
  unsigned short* Wc = (unsigned short*)carve(128 * 128 * 2);
  unsigned short* Wa = (unsigned short*)carve(256 * 128 * 2);         // att wt, 256-row pad
  unsigned short* Wr = (unsigned short*)carve(128 * 384 * 2);
  float* attbp = (float*)carve(256 * 4);
  float* Mg    = (float*)carve(2601 * 4);       // M(2500)+usum(50)+vt1(50)+norm(1)
  float* usum  = Mg + 2500; float* vt1 = Mg + 2550; float* normv = Mg + 2600;
  float* bns   = (float*)carve(256 * 4); float* bnq = bns + 128;

  const int rblk = Nn / 32;                     // 3125 blocks of 2 row-tiles
  const float invN = 1.0f / (float)Nn;
  auto g1 = [](long long n) { return (unsigned)((n + 255) / 256); };

  zpad<<<g1((long long)Nn * 28), 256, 0, stream>>>(Ac, Nn);

  const float* xin = x0;
  for (int L = 0; L < 3; ++L) {
    int Mout = (L == 2) ? 64 : 128;
    // --- weight prep (bf16, transposed, padded) ---
    prep_wt<<<g1(128 * 128), 256, 0, stream>>>(conv_w[L], 128, 128, Wc, 128, 128);
    prep_wt<<<g1(256 * 128), 256, 0, stream>>>(att_w[L], 128, 200, Wa, 128, 256);
    padb<<<1, 256, 0, stream>>>(att_b[L], 200, attbp, 256);
    prep_wt<<<g1(Mout * 384), 256, 0, stream>>>(red_w[L], 356, Mout, Wr, 384, Mout);
    // --- stage x as bf16 A-matrix ---
    cvt_cols<<<g1((long long)Nn * 128), 256, 0, stream>>>(xin, 128, Axb, 128, 0, 128, Nn);
    // --- GEMMs (WMMA, 1x4 blocked) ---
    gemm_bf16_wmma4<<<dim3(rblk, 2), 64, 0, stream>>>(Axb, Wc, nullptr, hbuf,
                                                      128, 128, 8, 128, 0);
    gemm_bf16_wmma4<<<dim3(rblk, 4), 64, 0, stream>>>(Axb, Wa, attbp, tbuf,
                                                      128, 128, 13, 208, 1);
    // --- edge aggregation ---
    fzero<<<g1((long long)Nn * 128), 256, 0, stream>>>(agg, (long long)Nn * 128);
    scatter_edges<<<g1(Ee * 32), 256, 0, stream>>>(hbuf, ei, ei + Ee, ew, agg, (int)Ee);
    // --- LRGA ---
    fzero<<<g1(2601), 256, 0, stream>>>(Mg, 2601);
    lrga_part<<<(Nn + 63) / 64, 256, 0, stream>>>(tbuf, 208, Mg, usum, vt1, Nn);
    norm_k<<<1, 64, 0, stream>>>(usum, vt1, normv, invN);
    // --- build bf16 concat [res | T | x_local | x] ---
    res_k<<<g1((long long)Nn * 64), 256, 0, stream>>>(tbuf, 208, Mg, normv, Ac, Nn);
    cvt_cols<<<g1((long long)Nn * 50), 256, 0, stream>>>(tbuf + 150, 208, Ac, 384, 50, 50, Nn);
    agg_fin<<<g1((long long)Nn * 128), 256, 0, stream>>>(agg, conv_b[L], Ac, Nn);
    cvt_cols<<<g1((long long)Nn * 128), 256, 0, stream>>>(xin, 128, Ac, 384, 228, 128, Nn);
    // --- reduction GEMM ---
    if (L < 2) {
      gemm_bf16_wmma4<<<dim3(rblk, 2), 64, 0, stream>>>(Ac, Wr, red_b[L], xbuf,
                                                        384, 384, 8, 128, 1);
      fzero<<<1, 256, 0, stream>>>(bns, 256);
      bn_stats<<<(Nn + 255) / 256, 256, 0, stream>>>(xbuf, bns, bnq, Nn);
      bn_apply<<<g1((long long)Nn * 128), 256, 0, stream>>>(xbuf, bns, bnq,
                                                            bn_g[L], bn_b[L], Nn, invN);
      xin = xbuf;
    } else {
      gemm_bf16_wmma4<<<dim3(rblk, 1), 64, 0, stream>>>(Ac, Wr, red_b[2],
                                                        (float*)d_out,
                                                        384, 384, 4, 64, 0);
    }
  }
}
